// GCNIIppi_82961588289743
// MI455X (gfx1250) — matrix operand order repaired and checked by reading, not compiled, not run on hardware
//
#include <hip/hip_runtime.h>
#include <hip/hip_bf16.h>
#include <math.h>

#define NN    50000
#define NFEAT 512
#define NHID  128
#define NE    800000
#define NL    8
#define NMUT  64

typedef __attribute__((ext_vector_type(16))) __bf16 v16bf;
typedef __attribute__((ext_vector_type(8)))  __bf16 v8bf;
typedef __attribute__((ext_vector_type(8)))  float  v8f;

#define LDS_STRIDE 40   // bf16 elems: 80 B rows -> 16-B aligned vector loads

// pack two fp32 -> bf16 pair, single 32-bit LDS store (v_cvt_pk_bf16_f32)
__device__ __forceinline__ void store_pk(__bf16* dst, float a, float b)
{
    union { __bf16 h[2]; unsigned u; } p;
    p.h[0] = (__bf16)a;
    p.h[1] = (__bf16)b;
    *(unsigned*)dst = p.u;
}

// A fragment (16x32 bf16, ISA half-lane layout): two contiguous 16-B runs
__device__ __forceinline__ v16bf load_frag_a(const __bf16* rowPtr, int half)
{
    v8bf lo = *(const v8bf*)(rowPtr + half * 8);
    v8bf hi = *(const v8bf*)(rowPtr + 16 + half * 8);
    return __builtin_shufflevector(lo, hi, 0, 1, 2, 3, 4, 5, 6, 7,
                                           8, 9, 10, 11, 12, 13, 14, 15);
}

// B fragment (32x16 bf16, column per lane, same K layout as A): two 16-B runs
__device__ __forceinline__ v16bf load_frag_b(const __bf16* colPtr, int half)
{
    v8bf lo = *(const v8bf*)(colPtr + half * 8);
    v8bf hi = *(const v8bf*)(colPtr + 16 + half * 8);
    return __builtin_shufflevector(lo, hi, 0, 1, 2, 3, 4, 5, 6, 7,
                                           8, 9, 10, 11, 12, 13, 14, 15);
}

// branchless row clamp: out-of-range rows (tail block only) load row NN-1;
// the waves that would consume those LDS rows are inactive, so staging can
// stay unguarded and the loads issue as one clause (no per-element waits).
__device__ __forceinline__ int clamp_row(int r) { return r < NN ? r : NN - 1; }

// ---------------------------------------------------------------------------
// fc0: out = relu(X @ W + b) written to both h0 and h.  128x128 tile/block,
// 8 waves, wave w owns rows [16w,16w+16), 8 col-tiles; K chunks of 32 in LDS.
// ---------------------------------------------------------------------------
__global__ __launch_bounds__(256)
void fc0_kernel(const float* __restrict__ x, const float* __restrict__ wx,
                const float* __restrict__ W, const float* __restrict__ bias,
                float* __restrict__ h0, float* __restrict__ wh0,
                float* __restrict__ h,  float* __restrict__ wh)
{
    const float* in  = (blockIdx.y == 0) ? x  : wx;
    float* out0      = (blockIdx.y == 0) ? h0 : wh0;
    float* out1      = (blockIdx.y == 0) ? h  : wh;

    const int rowBase = blockIdx.x * 128;
    const int tid  = threadIdx.x;
    const int lane = tid & 31;
    const int wave = tid >> 5;
    const int half = lane >> 4;
    const int l15  = lane & 15;

    __shared__ __align__(16) __bf16 As[128][LDS_STRIDE];  // [row][k]
    __shared__ __align__(16) __bf16 Bs[128][LDS_STRIDE];  // [col][k]

    v8f z = {};
    v8f acc[8];
#pragma unroll
    for (int c = 0; c < 8; ++c) acc[c] = z;

    const int  mrow       = wave * 16 + l15;
    const bool waveActive = (rowBase + wave * 16) < NN;

    for (int kc = 0; kc < NFEAT / 32; ++kc) {
        // ---- stage A chunk: float2 loads -> packed bf16x2 stores, branchless
#pragma unroll
        for (int i = 0; i < 8; ++i) {
            int e  = i * 256 + tid;          // pair index, 2048 pairs
            int r  = e >> 4;                 // 16 pairs per 32-wide row
            int k  = (e & 15) * 2;
            int gr = clamp_row(rowBase + r);
            float2 f = *(const float2*)&in[(size_t)gr * NFEAT + kc * 32 + k];
            store_pk(&As[r][k], f.x, f.y);
        }
        // ---- stage B chunk transposed: Bs[n][k] = W[(kc*32+k)*NHID + n]
#pragma unroll
        for (int i = 0; i < 8; ++i) {
            int p  = i * 256 + tid;          // 2048 pairs
            int n  = p & 127;
            int k0 = (p >> 7) * 2;
            float b0 = W[(size_t)(kc * 32 + k0) * NHID + n];
            float b1 = W[(size_t)(kc * 32 + k0 + 1) * NHID + n];
            store_pk(&Bs[n][k0], b0, b1);
        }
        // prefetch next A chunk (fc0 streams 2x102 MB of HBM -> bound here)
        if (kc + 1 < NFEAT / 32) {
            int pr = clamp_row(rowBase + (tid >> 1));
            __builtin_prefetch(&in[(size_t)pr * NFEAT + (kc + 1) * 32 +
                                   (tid & 1) * 16], 0, 3);
        }
        __syncthreads();

        if (waveActive) {
            v16bf a = load_frag_a(&As[mrow][0], half);
#pragma unroll
            for (int c = 0; c < 8; ++c) {
                v16bf bb = load_frag_b(&Bs[c * 16 + l15][0], half);
                acc[c] = __builtin_amdgcn_wmma_f32_16x16x32_bf16(
                    false, a, false, bb, (short)0, acc[c], false, false);
            }
        }
        __syncthreads();
    }

    if (waveActive) {
#pragma unroll
        for (int c = 0; c < 8; ++c) {
            int gcol = c * 16 + l15;
            float bv = bias[gcol];
#pragma unroll
            for (int r = 0; r < 8; ++r) {
                int grow = rowBase + wave * 16 + r + half * 8;  // C layout
                float v = acc[c][r] + bv;
                v = v > 0.f ? v : 0.f;
                size_t idx = (size_t)grow * NHID + gcol;
                out0[idx] = v;
                out1[idx] = v;
            }
        }
    }
}

// ---------------------------------------------------------------------------
// layer GEMM: h = relu(theta*(sup @ W) + (1-theta)*sup + h),
//             sup = 0.9*hi + 0.1*h0.  K = 128 (4 chunks of 32).
// ---------------------------------------------------------------------------
__global__ __launch_bounds__(256)
void layer_gemm(const float* __restrict__ hi, const float* __restrict__ h0,
                float* __restrict__ h, const float* __restrict__ W,
                float theta)
{
    const int rowBase = blockIdx.x * 128;
    const int tid  = threadIdx.x;
    const int lane = tid & 31;
    const int wave = tid >> 5;
    const int half = lane >> 4;
    const int l15  = lane & 15;

    __shared__ __align__(16) __bf16 As[128][LDS_STRIDE];
    __shared__ __align__(16) __bf16 Bs[128][LDS_STRIDE];

    v8f z = {};
    v8f acc[8];
#pragma unroll
    for (int c = 0; c < 8; ++c) acc[c] = z;

    const int  mrow       = wave * 16 + l15;
    const bool waveActive = (rowBase + wave * 16) < NN;

    for (int kc = 0; kc < NHID / 32; ++kc) {
        // ---- stage A = sup = 0.9*hi + 0.1*h0 (branchless, clause-friendly)
#pragma unroll
        for (int i = 0; i < 8; ++i) {
            int e  = i * 256 + tid;
            int r  = e >> 4;
            int k  = (e & 15) * 2;
            int gr = clamp_row(rowBase + r);
            size_t idx = (size_t)gr * NHID + kc * 32 + k;
            float2 a = *(const float2*)&hi[idx];
            float2 b = *(const float2*)&h0[idx];
            store_pk(&As[r][k], 0.9f * a.x + 0.1f * b.x,
                                0.9f * a.y + 0.1f * b.y);
        }
#pragma unroll
        for (int i = 0; i < 8; ++i) {
            int p  = i * 256 + tid;
            int n  = p & 127;
            int k0 = (p >> 7) * 2;
            float b0 = W[(size_t)(kc * 32 + k0) * NHID + n];
            float b1 = W[(size_t)(kc * 32 + k0 + 1) * NHID + n];
            store_pk(&Bs[n][k0], b0, b1);
        }
        __syncthreads();

        if (waveActive) {
            v16bf a = load_frag_a(&As[mrow][0], half);
#pragma unroll
            for (int c = 0; c < 8; ++c) {
                v16bf bb = load_frag_b(&Bs[c * 16 + l15][0], half);
                acc[c] = __builtin_amdgcn_wmma_f32_16x16x32_bf16(
                    false, a, false, bb, (short)0, acc[c], false, false);
            }
        }
        __syncthreads();
    }

    if (waveActive) {
        const float omt = 1.0f - theta;
#pragma unroll
        for (int c = 0; c < 8; ++c) {
            int gcol = c * 16 + l15;
#pragma unroll
            for (int r = 0; r < 8; ++r) {
                int grow = rowBase + wave * 16 + r + half * 8;
                size_t idx = (size_t)grow * NHID + gcol;
                float sup = 0.9f * hi[idx] + 0.1f * h0[idx];
                float v = theta * acc[c][r] + omt * sup + h[idx];
                h[idx] = v > 0.f ? v : 0.f;
            }
        }
    }
}

// ---------------------------------------------------------------------------
// SPMM scatter: hi[dst] += val * h[src].  One wave per edge, float4 per lane.
// h rows (25.6 MB) are L2-resident; atomics resolve at L2 (no-return ->
// STOREcnt, no wait stalls).
// ---------------------------------------------------------------------------
__global__ __launch_bounds__(256)
void spmm_kernel(const int* __restrict__ src, const int* __restrict__ dst,
                 const float* __restrict__ val, const float* __restrict__ h,
                 float* __restrict__ hi)
{
    int wave = threadIdx.x >> 5, lane = threadIdx.x & 31;
    int e = blockIdx.x * 8 + wave;
    if (e >= NE) return;
    int s = src[e], d = dst[e];
    float v = val[e];
    const float4* hs = (const float4*)(h + (size_t)s * NHID);
    float4 hv = hs[lane];
    float* out = hi + (size_t)d * NHID + lane * 4;
    atomicAdd(out + 0, v * hv.x);
    atomicAdd(out + 1, v * hv.y);
    atomicAdd(out + 2, v * hv.z);
    atomicAdd(out + 3, v * hv.w);
}

__global__ __launch_bounds__(256)
void zero_kernel(float* __restrict__ p, int n)
{
    int i = blockIdx.x * 256 + threadIdx.x;
    if (i < n) p[i] = 0.0f;
}

// ---------------------------------------------------------------------------
// Final head: differ_sum over mutation sites -> fc -> relu -> concat(aux) -> fc2
// d_out = [out(1), d(4)]
// ---------------------------------------------------------------------------
__global__ __launch_bounds__(128)
void final_kernel(const float* __restrict__ h, const float* __restrict__ wh,
                  const int* __restrict__ mut, const float* __restrict__ aux,
                  const float* __restrict__ fcW, const float* __restrict__ fcb,
                  const float* __restrict__ fc2W, const float* __restrict__ fc2b,
                  float* __restrict__ out)
{
    __shared__ float ds[NHID];
    __shared__ float dd[4];
    int f = threadIdx.x;
    float s = 0.f;
    for (int m = 0; m < NMUT; ++m) {
        int site = mut[m];
        s += h[(size_t)site * NHID + f] - wh[(size_t)site * NHID + f];
    }
    ds[f] = s;
    __syncthreads();
    if (f < 4) {
        float d = fcb[f];
        for (int k = 0; k < NHID; ++k) d += ds[k] * fcW[k * 4 + f];
        out[1 + f] = d;                      // tuple element "d"
        dd[f] = d > 0.f ? d : 0.f;
    }
    __syncthreads();
    if (f == 0) {
        float a = fc2b[0];
        for (int j = 0; j < 4; ++j) a += dd[j] * fc2W[j];
        a += aux[4] * (float)NMUT * fc2W[4]; // aux[-4:-2] * len(mutation_site)
        a += aux[5] * (float)NMUT * fc2W[5];
        out[0] = a;                          // tuple element "out"
    }
}

// ---------------------------------------------------------------------------
extern "C" void kernel_launch(void* const* d_in, const int* in_sizes, int n_in,
                              void* d_out, int out_size, void* d_ws, size_t ws_size,
                              hipStream_t stream)
{
    (void)in_sizes; (void)n_in; (void)out_size; (void)ws_size;

    const float* x      = (const float*)d_in[0];
    const float* wx     = (const float*)d_in[1];
    const int*   esrc   = (const int*)  d_in[2];
    const int*   edst   = (const int*)  d_in[3];
    const float* eval_  = (const float*)d_in[4];
    const int*   wesrc  = (const int*)  d_in[5];
    const int*   wedst  = (const int*)  d_in[6];
    const float* weval  = (const float*)d_in[7];
    const int*   mut    = (const int*)  d_in[8];
    const float* aux    = (const float*)d_in[9];
    /* d_in[10] = nodes scalar (compile-time constant here) */
    const float* fc0W   = (const float*)d_in[11];
    const float* fc0b   = (const float*)d_in[12];
    const float* convW  = (const float*)d_in[13];
    const float* fcW    = (const float*)d_in[14];
    const float* fcb    = (const float*)d_in[15];
    const float* fc2W   = (const float*)d_in[16];
    const float* fc2b   = (const float*)d_in[17];
    float* out = (float*)d_out;

    const size_t NM = (size_t)NN * NHID;
    float* h   = (float*)d_ws;
    float* wh  = h   + NM;
    float* h0  = wh  + NM;
    float* wh0 = h0  + NM;
    float* hi  = wh0 + NM;

    const int nRowBlocks = (NN + 127) / 128;          // 391
    const int spmmBlocks = (NE + 7) / 8;              // 100000
    const int zeroBlocks = ((int)NM + 255) / 256;     // 25000

    fc0_kernel<<<dim3(nRowBlocks, 2), 256, 0, stream>>>(
        x, wx, fc0W, fc0b, h0, wh0, h, wh);

    for (int i = 0; i < NL; ++i) {
        float theta = logf(0.5f / (float)(i + 1) + 1.0f);
        const float* Wl = convW + (size_t)i * NHID * NHID;

        // mutant graph
        zero_kernel<<<zeroBlocks, 256, 0, stream>>>(hi, (int)NM);
        spmm_kernel<<<spmmBlocks, 256, 0, stream>>>(esrc, edst, eval_, h, hi);
        layer_gemm<<<nRowBlocks, 256, 0, stream>>>(hi, h0, h, Wl, theta);

        // wild-type graph (shared weights)
        zero_kernel<<<zeroBlocks, 256, 0, stream>>>(hi, (int)NM);
        spmm_kernel<<<spmmBlocks, 256, 0, stream>>>(wesrc, wedst, weval, wh, hi);
        layer_gemm<<<nRowBlocks, 256, 0, stream>>>(hi, wh0, wh, Wl, theta);
    }

    final_kernel<<<1, 128, 0, stream>>>(h, wh, mut, aux, fcW, fcb, fc2W, fc2b, out);
}
